// GPT3DevAttention_7275674599474
// MI455X (gfx1250) — matrix operand order repaired and checked by reading, not compiled, not run on hardware
//
#include <hip/hip_runtime.h>

// ---------------------------------------------------------------------------
// GPT-3 style causal attention layer for MI455X (gfx1250, wave32, WMMA).
// bf16 operands into v_wmma_f32_16x16x32_bf16, f32 accumulation.
// Layout strategy: weights converted transposed (Wt[N][K]) and V stored
// transposed (Vt[B,H,DH,S]) so every WMMA fragment is a contiguous vector
// load -- no LDS scatter/gather anywhere.
// ---------------------------------------------------------------------------

typedef __attribute__((ext_vector_type(16))) __bf16 v16bf;
typedef __attribute__((ext_vector_type(8)))  __bf16 v8bf;
typedef __attribute__((ext_vector_type(8)))  float  v8f;

#define S_LEN  2048
#define DMODEL 1024
#define NH     16
#define DH     64
#define BATCH  2

union AV  { v16bf v; v8bf h[2]; };

static __device__ __forceinline__ v8f vzero8() {
    v8f z = {0.f, 0.f, 0.f, 0.f, 0.f, 0.f, 0.f, 0.f};
    return z;
}

static __device__ __forceinline__ float redmax16(float v) {
#pragma unroll
    for (int m = 1; m < 16; m <<= 1) v = fmaxf(v, __shfl_xor(v, m, 16));
    return v;
}
static __device__ __forceinline__ float redsum16(float v) {
#pragma unroll
    for (int m = 1; m < 16; m <<= 1) v += __shfl_xor(v, m, 16);
    return v;
}

// ---------------------------------------------------------------------------
// f32 -> bf16 conversion (bandwidth kernel)
// ---------------------------------------------------------------------------
__global__ void cvt_f32_to_bf16(const float* __restrict__ src,
                                __bf16* __restrict__ dst, int n) {
    int i      = blockIdx.x * blockDim.x + threadIdx.x;
    int stride = gridDim.x * blockDim.x;
    for (; i < n; i += stride) dst[i] = (__bf16)src[i];
}

// f32 [1024][N] -> bf16 transposed [N][1024] (one-time, L2-resident)
__global__ void cvt_transpose_bf16(const float* __restrict__ src,
                                   __bf16* __restrict__ dst, int N) {
    int i      = blockIdx.x * blockDim.x + threadIdx.x;
    int stride = gridDim.x * blockDim.x;
    const int total = N * 1024;
    for (; i < total; i += stride) {
        const int k = i & 1023;      // dst fast dim
        const int n = i >> 10;
        dst[i] = (__bf16)src[(size_t)k * N + n];
    }
}

// ---------------------------------------------------------------------------
// WMMA GEMM core: C[128x128] tile. A[M][1024] bf16 row-major,
// Wt[N][1024] bf16 (pre-transposed). 8 waves x (64x32) = 4x2 WMMA tiles.
// LDS double-buffered, one barrier per k-step.
// ---------------------------------------------------------------------------
__device__ __forceinline__ void gemm_core(const __bf16* __restrict__ A,
                                          const __bf16* __restrict__ Wt,
                                          int rowBase, int nBase,
                                          v8f (&acc)[4][2]) {
    __shared__ __align__(32) __bf16 As[2][128][48];   // [buf][row][k]
    __shared__ __align__(32) __bf16 Bs[2][128][48];   // [buf][col][k]

    const int tid  = threadIdx.x;
    const int lane = tid & 31;
    const int w    = tid >> 5;
    const int half = lane >> 4;
    const int col  = lane & 15;
    const int wm   = (w >> 2) * 64;   // wave row offset: 0 / 64
    const int wn   = (w & 3) * 32;    // wave col offset: 0..96

    const int r  = tid >> 1;          // staged row/col index: 0..127
    const int kc = (tid & 1) * 16;    // k sub-chunk: 0 / 16
    const __bf16* gA = A  + (size_t)(rowBase + r) * 1024 + kc;
    const __bf16* gB = Wt + (size_t)(nBase  + r) * 1024 + kc;

#pragma unroll
    for (int rt = 0; rt < 4; ++rt)
#pragma unroll
        for (int ct = 0; ct < 2; ++ct) acc[rt][ct] = vzero8();

    v16bf ra = *(const v16bf*)gA;     // tile 0 in flight
    v16bf rb = *(const v16bf*)gB;

    int buf = 0;
    for (int k0 = 0; k0 < 1024; k0 += 32, buf ^= 1) {
        *(v16bf*)&As[buf][r][kc] = ra;
        *(v16bf*)&Bs[buf][r][kc] = rb;
        __syncthreads();              // s_wait_dscnt 0 + barrier

        if (k0 + 32 < 1024) {         // issue next tile loads under the WMMAs
            ra = *(const v16bf*)(gA + k0 + 32);
            rb = *(const v16bf*)(gB + k0 + 32);
        }

        v16bf af[4];
#pragma unroll
        for (int rt = 0; rt < 4; ++rt) {
            const int row = wm + rt * 16 + col;
            AV t;
            t.h[0] = *(const v8bf*)&As[buf][row][half * 16];      // K 0-7 / 16-23
            t.h[1] = *(const v8bf*)&As[buf][row][half * 16 + 8];  // K 8-15 / 24-31
            af[rt] = t.v;
        }
        v16bf bfm[2];
#pragma unroll
        for (int ct = 0; ct < 2; ++ct) {
            const int c = wn + ct * 16 + col;
            bfm[ct] = *(const v16bf*)&Bs[buf][c][half * 16];      // K 0-15 / 16-31
        }
#pragma unroll
        for (int rt = 0; rt < 4; ++rt)
#pragma unroll
            for (int ct = 0; ct < 2; ++ct)
                acc[rt][ct] = __builtin_amdgcn_wmma_f32_16x16x32_bf16(
                    false, af[rt], false, bfm[ct], (short)0, acc[rt][ct],
                    false, false);
    }
}

// ---------------------------------------------------------------------------
// QKV projection: qkv = hidden @ Wqkv + bqkv.
// Q -> [B,H,S,DH]*0.125, K -> [B,H,S,DH], V -> transposed [B,H,DH,S].
// ---------------------------------------------------------------------------
__global__ __launch_bounds__(256)
void gemm_qkv(const __bf16* __restrict__ A, const __bf16* __restrict__ Wt,
              const float* __restrict__ bias, __bf16* __restrict__ Qo,
              __bf16* __restrict__ Ko, __bf16* __restrict__ Vt) {
    const int lane = threadIdx.x & 31;
    const int w    = threadIdx.x >> 5;
    const int half = lane >> 4, col = lane & 15;
    const int wm   = (w >> 2) * 64, wn = (w & 3) * 32;
    const int rowBase = blockIdx.y * 128;
    const int nBase   = blockIdx.x * 128;

    v8f acc[4][2];
    gemm_core(A, Wt, rowBase, nBase, acc);

#pragma unroll
    for (int rt = 0; rt < 4; ++rt)
#pragma unroll
        for (int ct = 0; ct < 2; ++ct) {
            const int gn  = nBase + wn + ct * 16 + col;
            const float b = bias[gn];
            const int sec = gn >> 10;          // 0:Q 1:K 2:V
            const int id  = gn & 1023;
            const int h   = id >> 6, dh = id & 63;
#pragma unroll
            for (int r = 0; r < 8; ++r) {
                const int gm = rowBase + wm + rt * 16 + r + half * 8;
                const int bb = gm >> 11, s = gm & 2047;
                const float v = acc[rt][ct][r] + b;
                const size_t bh = (size_t)(bb * NH + h);
                if (sec == 0)
                    Qo[(bh * S_LEN + s) * DH + dh] = (__bf16)(v * 0.125f);
                else if (sec == 1)
                    Ko[(bh * S_LEN + s) * DH + dh] = (__bf16)v;
                else
                    Vt[(bh * DH + dh) * S_LEN + s] = (__bf16)v;
            }
        }
}

// ---------------------------------------------------------------------------
// Output projection: out = attn @ Wproj + bproj (f32 out)
// ---------------------------------------------------------------------------
__global__ __launch_bounds__(256)
void gemm_proj(const __bf16* __restrict__ A, const __bf16* __restrict__ Wt,
               const float* __restrict__ bias, float* __restrict__ out) {
    const int lane = threadIdx.x & 31;
    const int w    = threadIdx.x >> 5;
    const int half = lane >> 4, col = lane & 15;
    const int wm   = (w >> 2) * 64, wn = (w & 3) * 32;
    const int rowBase = blockIdx.y * 128;
    const int nBase   = blockIdx.x * 128;

    v8f acc[4][2];
    gemm_core(A, Wt, rowBase, nBase, acc);

#pragma unroll
    for (int rt = 0; rt < 4; ++rt)
#pragma unroll
        for (int ct = 0; ct < 2; ++ct) {
            const int gn  = nBase + wn + ct * 16 + col;
            const float b = bias[gn];
#pragma unroll
            for (int r = 0; r < 8; ++r) {
                const int gm = rowBase + wm + rt * 16 + r + half * 8;
                out[(size_t)gm * DMODEL + gn] = acc[rt][ct][r] + b;
            }
        }
}

// ---------------------------------------------------------------------------
// Flash attention: each wave owns 16 query rows of one (b,h); streams 32-key
// tiles with online softmax. Q pre-scaled; K [B,H,S,DH]; V transposed
// [B,H,DH,S] so K^T and V B-fragments are single 32B global loads.
// grid = (S/128, B*H), block = 256 (8 waves).
// ---------------------------------------------------------------------------
__global__ __launch_bounds__(256)
void flash_attn(const __bf16* __restrict__ Q, const __bf16* __restrict__ K,
                const __bf16* __restrict__ Vt, __bf16* __restrict__ O) {
    __shared__ __align__(32) __bf16 Ps[8][16][40];  // per-wave P tile [q][key]

    const int tid  = threadIdx.x;
    const int lane = tid & 31;
    const int w    = tid >> 5;
    const int half = lane >> 4, col = lane & 15;
    const int bh    = blockIdx.y;                   // b*NH + h
    const int qBase = blockIdx.x * 128 + w * 16;

    const __bf16* Qb  = Q  + (size_t)bh * S_LEN * DH;
    const __bf16* Kb  = K  + (size_t)bh * S_LEN * DH;
    const __bf16* Vtb = Vt + (size_t)bh * DH * S_LEN;

    // Q A-fragments (dh 0-31 and 32-63), loaded once.
    v16bf aq[2];
    {
        const int qr = qBase + col;
#pragma unroll
        for (int dd = 0; dd < 2; ++dd) {
            AV t;
            t.h[0] = *(const v8bf*)(Qb + (size_t)qr * DH + dd * 32 + half * 16);
            t.h[1] = *(const v8bf*)(Qb + (size_t)qr * DH + dd * 32 + half * 16 + 8);
            aq[dd] = t.v;
        }
    }

    v8f accO[4];
    float m[8], l[8];
#pragma unroll
    for (int d = 0; d < 4; ++d) accO[d] = vzero8();
#pragma unroll
    for (int r = 0; r < 8; ++r) { m[r] = -3.0e38f; l[r] = 0.0f; }

    const int nkt = (qBase + 15) / 32 + 1;   // causal key-tile limit
    for (int kt = 0; kt < nkt; ++kt) {
        const int kb = kt * 32;

        // ---- scores: S = (Q*scale) K^T, two 16-col tiles, K from global ----
        v16bf bk[2][2];
#pragma unroll
        for (int kc = 0; kc < 2; ++kc)
#pragma unroll
            for (int dd = 0; dd < 2; ++dd)
                bk[kc][dd] = *(const v16bf*)(Kb +
                    (size_t)(kb + kc * 16 + col) * DH + dd * 32 + half * 16);

        v8f c0 = vzero8(), c1 = vzero8();
        c0 = __builtin_amdgcn_wmma_f32_16x16x32_bf16(false, aq[0], false, bk[0][0], (short)0, c0, false, false);
        c0 = __builtin_amdgcn_wmma_f32_16x16x32_bf16(false, aq[1], false, bk[0][1], (short)0, c0, false, false);
        c1 = __builtin_amdgcn_wmma_f32_16x16x32_bf16(false, aq[0], false, bk[1][0], (short)0, c1, false, false);
        c1 = __builtin_amdgcn_wmma_f32_16x16x32_bf16(false, aq[1], false, bk[1][1], (short)0, c1, false, false);

        // ---- V B-fragments for this key tile (contiguous along S) ----
        v16bf bv[4];
#pragma unroll
        for (int d = 0; d < 4; ++d)
            bv[d] = *(const v16bf*)(Vtb +
                (size_t)(d * 16 + col) * S_LEN + kb + half * 16);

        // ---- causal mask + online softmax (rows = r + 8*half) ----
#pragma unroll
        for (int r = 0; r < 8; ++r) {
            const int qrow = qBase + r + half * 8;
            const float s0 = (kb + col      <= qrow) ? c0[r] : -1.0e30f;
            const float s1 = (kb + 16 + col <= qrow) ? c1[r] : -1.0e30f;
            const float mt = redmax16(fmaxf(s0, s1));
            const float mn = fmaxf(m[r], mt);
            const float alpha = __expf(m[r] - mn);
            const float p0 = __expf(s0 - mn);
            const float p1 = __expf(s1 - mn);
            l[r] = l[r] * alpha + redsum16(p0 + p1);
            m[r] = mn;
#pragma unroll
            for (int d = 0; d < 4; ++d) accO[d][r] *= alpha;
            Ps[w][r + half * 8][col]      = (__bf16)p0;
            Ps[w][r + half * 8][16 + col] = (__bf16)p1;
        }

        // Same-wave LDS producer->consumer ordering (split CDNA5 counter).
        asm volatile("s_wait_dscnt 0" ::: "memory");

        // ---- P A-fragment (C-layout -> A-layout via LDS) ----
        AV ap;
        ap.h[0] = *(const v8bf*)&Ps[w][col][half * 16];
        ap.h[1] = *(const v8bf*)&Ps[w][col][half * 16 + 8];

        // ---- O += P @ V : 4 dh tiles ----
#pragma unroll
        for (int d = 0; d < 4; ++d)
            accO[d] = __builtin_amdgcn_wmma_f32_16x16x32_bf16(
                false, ap.v, false, bv[d], (short)0, accO[d], false, false);
    }

    // ---- normalize and store attn [B, S, H*DH] bf16 ----
    const int bb = bh >> 4, h = bh & 15;
#pragma unroll
    for (int d = 0; d < 4; ++d)
#pragma unroll
        for (int r = 0; r < 8; ++r) {
            const int qrow = qBase + r + half * 8;
            const float o  = accO[d][r] / l[r];
            const size_t off = ((size_t)bb * S_LEN + qrow) * DMODEL +
                               h * DH + d * 16 + col;
            O[off] = (__bf16)o;
        }
}

// ---------------------------------------------------------------------------
// Host launcher
// ---------------------------------------------------------------------------
extern "C" void kernel_launch(void* const* d_in, const int* in_sizes, int n_in,
                              void* d_out, int out_size, void* d_ws, size_t ws_size,
                              hipStream_t stream) {
    const float* hs    = (const float*)d_in[0];
    // d_in[1] = attention_mask: exact causal additive mask, applied analytically.
    const float* Wqkv  = (const float*)d_in[2];
    const float* bqkv  = (const float*)d_in[3];
    const float* Wproj = (const float*)d_in[4];
    const float* bproj = (const float*)d_in[5];
    float* out = (float*)d_out;

    char* ws = (char*)d_ws;
    size_t off = 0;
    auto carve = [&](size_t elems) {
        void* p = ws + off;
        off = (off + elems * sizeof(__bf16) + 255) & ~(size_t)255;
        return (__bf16*)p;
    };
    const size_t nHid  = (size_t)BATCH * S_LEN * DMODEL;      // 4.19M
    const size_t nWqkv = (size_t)DMODEL * 3 * DMODEL;         // 3.15M
    const size_t nWprj = (size_t)DMODEL * DMODEL;             // 1.05M
    const size_t nQKV  = (size_t)BATCH * NH * S_LEN * DH;     // 4.19M

    __bf16* hsb  = carve(nHid);
    __bf16* wqt  = carve(nWqkv);    // Wqkv transposed [3072][1024]
    __bf16* wpt  = carve(nWprj);    // Wproj transposed [1024][1024]
    __bf16* Qb   = carve(nQKV);     // [B,H,S,DH]
    __bf16* Kb   = carve(nQKV);     // [B,H,S,DH]
    __bf16* Vtb  = carve(nQKV);     // [B,H,DH,S]
    __bf16* attn = carve(nHid);     // [B,S,H*DH]; ~50 MB workspace total

    cvt_f32_to_bf16<<<2048, 256, 0, stream>>>(hs, hsb, (int)nHid);
    cvt_transpose_bf16<<<2048, 256, 0, stream>>>(Wqkv, wqt, 3 * DMODEL);
    cvt_transpose_bf16<<<1024, 256, 0, stream>>>(Wproj, wpt, DMODEL);

    // QKV projection: M=4096, N=3072
    gemm_qkv<<<dim3(24, 32), 256, 0, stream>>>(hsb, wqt, bqkv, Qb, Kb, Vtb);

    // Flash attention: (S/128 q-blocks, B*H)
    flash_attn<<<dim3(16, 32), 256, 0, stream>>>(Qb, Kb, Vtb, attn);

    // Output projection: M=4096, N=1024
    gemm_proj<<<dim3(8, 32), 256, 0, stream>>>(attn, wpt, bproj, out);
}
